// MixtralAttention_9371618640147
// MI455X (gfx1250) — matrix operand order repaired and checked by reading, not compiled, not run on hardware
//
#include <hip/hip_runtime.h>
#include <hip/hip_bf16.h>
#include <stdint.h>

// Problem constants (Mixtral attention layer)
#define B_   2
#define S_   1024
#define H_   4096
#define NH_  32
#define NKV_ 8
#define D_   128
#define GROUPS_ (NH_/NKV_)
#define SCALE_ 0.08838834764831845f   // D^-0.5

typedef __bf16 bf16_t;
typedef bf16_t v16bf __attribute__((ext_vector_type(16)));
typedef float  v8f   __attribute__((ext_vector_type(8)));

struct alignas(16) B16 { unsigned int w[4]; };   // 16-byte chunk (8 bf16)

union FragBF {
    v16bf v;
    B16 q[2];
    unsigned short u[16];
};

// ---- Tensor Data Mover support (guarded; falls back to vector copies) ----
#if defined(__has_builtin)
# if __has_builtin(__builtin_amdgcn_tensor_load_to_lds) && \
     __has_builtin(__builtin_amdgcn_s_wait_tensorcnt)
#  define HAVE_TDM 1
# endif
#endif
#ifndef HAVE_TDM
# define HAVE_TDM 0
#endif

#if HAVE_TDM
typedef unsigned int v4u_t __attribute__((ext_vector_type(4)));
typedef int v8i_t __attribute__((ext_vector_type(8)));
typedef int v4i_t __attribute__((ext_vector_type(4)));

// Generic pointer into the LDS aperture truncates to the LDS byte offset
__device__ __forceinline__ unsigned lds_off_of(const void* p) {
    return (unsigned)(unsigned long long)p;
}

// 2-D tile load Global->LDS via TDM (D# per CDNA5 ISA).
// All dims/strides in elements (data_size = 2 bytes). Tile rows are packed
// contiguously in LDS: LDS image = [tile_d1][tile_d0].
// This toolchain's builtin takes 6 args (g0, g1, g2, g3, g4, cpol).
__device__ __forceinline__ void tdm_load_2d(unsigned lds_off, const void* gptr,
                                            unsigned tensor_d0, unsigned tensor_d1,
                                            unsigned long long stride_d0,
                                            unsigned tile_d0, unsigned tile_d1) {
    unsigned long long ga = (unsigned long long)gptr;
    v4u_t g0;
    g0.x = 1u;                                   // count=1 (valid user descriptor)
    g0.y = lds_off;                              // lds_addr
    g0.z = (unsigned)ga;                         // global_addr[31:0]
    g0.w = (unsigned)((ga >> 32) & 0x01FFFFFFu)  // global_addr[56:32]
         | 0x80000000u;                          // type=2 ("image")
    v8i_t g1;
    g1[0] = (int)(1u << 16);                     // data_size=1 (2 bytes)
    g1[1] = (int)((tensor_d0 & 0xFFFFu) << 16);  // tensor_dim0[15:0] @ bits 63:48
    g1[2] = (int)(((tensor_d0 >> 16) & 0xFFFFu)  // tensor_dim0[31:16]
         | ((tensor_d1 & 0xFFFFu) << 16));       // tensor_dim1[15:0]
    g1[3] = (int)(((tensor_d1 >> 16) & 0xFFFFu)  // tensor_dim1[31:16]
         | ((tile_d0 & 0xFFFFu) << 16));         // tile_dim0
    g1[4] = (int)(tile_d1 & 0xFFFFu);            // tile_dim1 (tile_dim2=0)
    g1[5] = (int)(unsigned)(stride_d0 & 0xFFFFFFFFull);  // dim0_stride[31:0]
    g1[6] = (int)(unsigned)((stride_d0 >> 32) & 0xFFFFull); // dim0_stride[47:32]
    g1[7] = 0;
    v4i_t g2 = {0, 0, 0, 0};
    v4i_t g3 = {0, 0, 0, 0};
    v8i_t g4 = {0, 0, 0, 0, 0, 0, 0, 0};
    __builtin_amdgcn_tensor_load_to_lds(g0, g1, g2, g3, g4, 0);
}
#endif

__device__ __forceinline__ unsigned short f2bf(float f) {
    union { float f; unsigned u; } c; c.f = f;
    unsigned u = c.u;
    return (unsigned short)((u + 0x7FFFu + ((u >> 16) & 1u)) >> 16); // RNE
}
__device__ __forceinline__ float bf2f(unsigned short h) {
    union { unsigned u; float f; } c; c.u = ((unsigned)h) << 16;
    return c.f;
}
__device__ __forceinline__ v8f vzero8() {
    v8f z = {0.f,0.f,0.f,0.f,0.f,0.f,0.f,0.f};
    return z;
}

// ---------------------------------------------------------------------------
// fp32 -> bf16 conversion, 8 elements/thread
// ---------------------------------------------------------------------------
__global__ void cvt_bf16_v8(const float* __restrict__ src,
                            unsigned short* __restrict__ dst, int n8) {
    int i = blockIdx.x * blockDim.x + threadIdx.x;
    if (i >= n8) return;
    const float4* s = (const float4*)src + (size_t)i * 2;
    float4 a = s[0], b = s[1];
    union { B16 q; unsigned short u[8]; } o;
    o.u[0]=f2bf(a.x); o.u[1]=f2bf(a.y); o.u[2]=f2bf(a.z); o.u[3]=f2bf(a.w);
    o.u[4]=f2bf(b.x); o.u[5]=f2bf(b.y); o.u[6]=f2bf(b.z); o.u[7]=f2bf(b.w);
    ((B16*)dst)[i] = o.q;
}

// ---------------------------------------------------------------------------
// Fused fp32->bf16 convert + transpose: src [R][C] f32 -> dst [C][R] bf16.
// 32x32 LDS tile, coalesced both sides.  Block (32,8).
// ---------------------------------------------------------------------------
__global__ __launch_bounds__(256)
void cvt_transpose_kernel(const float* __restrict__ src,
                          unsigned short* __restrict__ dst, int R, int C) {
    __shared__ unsigned short tile[32][33];
    const int c0 = blockIdx.x * 32, r0 = blockIdx.y * 32;
    for (int i = threadIdx.y; i < 32; i += 8)
        tile[i][threadIdx.x] = f2bf(src[(size_t)(r0 + i) * C + c0 + threadIdx.x]);
    __syncthreads();
    for (int i = threadIdx.y; i < 32; i += 8)
        dst[(size_t)(c0 + i) * R + r0 + threadIdx.x] = tile[threadIdx.x][i];
}

// bf16 [R][C] -> bf16 [C][R]
__global__ __launch_bounds__(256)
void transpose_bf16_kernel(const unsigned short* __restrict__ src,
                           unsigned short* __restrict__ dst, int R, int C) {
    __shared__ unsigned short tile[32][33];
    const int c0 = blockIdx.x * 32, r0 = blockIdx.y * 32;
    for (int i = threadIdx.y; i < 32; i += 8)
        tile[i][threadIdx.x] = src[(size_t)(r0 + i) * C + c0 + threadIdx.x];
    __syncthreads();
    for (int i = threadIdx.y; i < 32; i += 8)
        dst[(size_t)(c0 + i) * R + r0 + threadIdx.x] = tile[threadIdx.x][i];
}

// ---------------------------------------------------------------------------
// bf16 GEMM: C[M,N] = A[M,K] * BT[N,K]^T.  A row-major, BT = B transposed
// (row-major in K), so every LDS store/load is a contiguous b128.
// Block: 128 threads = 4 waves; block tile 64(M) x 128(N); wave 32x64
// (2x4 WMMA accums).  K step 32.  Tiles staged by TDM when available.
// ---------------------------------------------------------------------------
template <bool OUT_BF16>
__global__ __launch_bounds__(128)
void gemm_bf16_kernel(const unsigned short* __restrict__ A,
                      const unsigned short* __restrict__ BT,
                      void* __restrict__ Cout,
                      int M, int N, int K) {
    __shared__ alignas(16) unsigned short ldsA[64 * 32];    // [m][k]
    __shared__ alignas(16) unsigned short ldsB[128 * 32];   // [n][k]

    const int tid  = threadIdx.x;
    const int lane = tid & 31;
    const int wave = tid >> 5;
    const int lh   = lane & 15;
    const int hi   = lane >> 4;

    const int m0 = blockIdx.y * 64;
    const int n0 = blockIdx.x * 128;
    const int wm = (wave >> 1) * 32;   // wave sub-tile row (2 waves in M)
    const int wn = (wave & 1) * 64;    // wave sub-tile col (2 waves in N)

    v8f acc[2][4];
    #pragma unroll
    for (int a = 0; a < 2; a++)
        #pragma unroll
        for (int b = 0; b < 4; b++) acc[a][b] = vzero8();

    for (int kk0 = 0; kk0 < K; kk0 += 32) {
#if HAVE_TDM
        if (wave == 0) {
            tdm_load_2d(lds_off_of(&ldsA[0]), A + (size_t)m0 * K + kk0,
                        (unsigned)K, (unsigned)M, (unsigned long long)K, 32, 64);
            tdm_load_2d(lds_off_of(&ldsB[0]), BT + (size_t)n0 * K + kk0,
                        (unsigned)K, (unsigned)N, (unsigned long long)K, 32, 128);
            __builtin_amdgcn_s_wait_tensorcnt(0);
        }
#else
        {   // A tile 64x32: thread covers half a row (2 x b128)
            const int arow = tid >> 1, acol = (tid & 1) * 16;
            const unsigned short* ga = A + (size_t)(m0 + arow) * K + kk0 + acol;
            *(B16*)&ldsA[arow * 32 + acol    ] = *(const B16*)(ga);
            *(B16*)&ldsA[arow * 32 + acol + 8] = *(const B16*)(ga + 8);
            // B tile 128x32: thread covers one BT row (4 x b128)
            const unsigned short* gb = BT + (size_t)(n0 + tid) * K + kk0;
            #pragma unroll
            for (int c = 0; c < 4; c++)
                *(B16*)&ldsB[tid * 32 + c * 8] = *(const B16*)(gb + c * 8);
        }
#endif
        __syncthreads();

        // ---- fragments (ISA 7.12.2 layouts), all contiguous b128 LDS loads ----
        FragBF fa[2], fb[4];
        #pragma unroll
        for (int ar = 0; ar < 2; ar++) {
            const unsigned short* p = &ldsA[(wm + ar * 16 + lh) * 32 + hi * 8];
            fa[ar].q[0] = *(const B16*)(p);
            fa[ar].q[1] = *(const B16*)(p + 16);
        }
        #pragma unroll
        for (int bc = 0; bc < 4; bc++) {
            const unsigned short* p = &ldsB[(wn + bc * 16 + lh) * 32 + hi * 8];
            fb[bc].q[0] = *(const B16*)(p);
            fb[bc].q[1] = *(const B16*)(p + 16);
        }
        #pragma unroll
        for (int ar = 0; ar < 2; ar++)
            #pragma unroll
            for (int bc = 0; bc < 4; bc++)
                acc[ar][bc] = __builtin_amdgcn_wmma_f32_16x16x32_bf16(
                    false, fa[ar].v, false, fb[bc].v,
                    (short)0, acc[ar][bc], false, false);
        __syncthreads();
    }

    // ---- epilogue: C layout, vgpr i -> row i (+8 for upper lanes) ----
    #pragma unroll
    for (int ar = 0; ar < 2; ar++) {
        #pragma unroll
        for (int bc = 0; bc < 4; bc++) {
            const int col = n0 + wn + bc * 16 + lh;
            const int rb  = m0 + wm + ar * 16 + hi * 8;
            #pragma unroll
            for (int i = 0; i < 8; i++) {
                float val = acc[ar][bc][i];
                if (OUT_BF16)
                    ((unsigned short*)Cout)[(size_t)(rb + i) * N + col] = f2bf(val);
                else
                    ((float*)Cout)[(size_t)(rb + i) * N + col] = val;
            }
        }
    }
}

// ---------------------------------------------------------------------------
// RoPE in place on bf16 tensor x[B,S,nh,D]; cos/sin are [B,S,D] fp32.
// ---------------------------------------------------------------------------
__global__ void rope_kernel(unsigned short* __restrict__ x,
                            const float* __restrict__ cosb,
                            const float* __restrict__ sinb, int nh) {
    int idx = blockIdx.x * blockDim.x + threadIdx.x;
    int total = B_ * S_ * nh * (D_ / 2);
    if (idx >= total) return;
    int d  = idx & 63;
    int h  = (idx >> 6) % nh;
    int bs = (idx >> 6) / nh;          // b*S + s
    unsigned short* row = x + ((size_t)bs * nh + h) * D_;
    float c  = cosb[(size_t)bs * D_ + d];
    float sn = sinb[(size_t)bs * D_ + d];
    float x1 = bf2f(row[d]);
    float x2 = bf2f(row[d + 64]);
    row[d]      = f2bf(x1 * c - x2 * sn);
    row[d + 64] = f2bf(x2 * c + x1 * sn);
}

// ---------------------------------------------------------------------------
// Flash attention, causal + key mask, GQA.  One wave per (b, h, 16 q rows).
// V is pre-transposed: VT[b][kvh*D+d][s], so each 32-key V chunk is a 2-D
// TDM tile (128 x 32) landing in LDS as [d][k] — B-fragments are b128 loads.
// ---------------------------------------------------------------------------
__global__ __launch_bounds__(32)
void flash_kernel(const unsigned short* __restrict__ Q,   // [B,S,NH,D] roped
                  const unsigned short* __restrict__ Kc,  // [B,S,NKV,D] roped
                  const unsigned short* __restrict__ VT,  // [B,NKV*D,S]
                  const unsigned char*  __restrict__ amask, // [B,S]
                  unsigned short* __restrict__ O) {        // [B,S,NH,D]
    __shared__ alignas(16) unsigned short ldsP[16 * 32];   // [m][k]
    __shared__ alignas(16) unsigned short ldsV[128 * 32];  // [d][k]

    const int lane = threadIdx.x & 31;
    const int lh = lane & 15;
    const int hi = lane >> 4;

    const int nqb = S_ / 16;
    const int qb = blockIdx.x % nqb;
    const int h  = (blockIdx.x / nqb) % NH_;
    const int b  = blockIdx.x / (nqb * NH_);
    const int kvh = h / GROUPS_;

    const size_t qstride = (size_t)NH_ * D_;
    const size_t kstride = (size_t)NKV_ * D_;
    const unsigned short* qbase  = Q  + (size_t)b * S_ * qstride + (size_t)h   * D_;
    const unsigned short* kbase  = Kc + (size_t)b * S_ * kstride + (size_t)kvh * D_;
    const unsigned short* vtbase = VT + ((size_t)b * NKV_ + kvh) * D_ * S_;

    // Q fragments for this 16-row tile: 4 A-frags covering D=128
    FragBF fq[4];
    {
        const unsigned short* qrow = qbase + (size_t)(qb * 16 + lh) * qstride;
        #pragma unroll
        for (int c = 0; c < 4; c++) {
            int d0 = c * 32 + hi * 8;
            fq[c].q[0] = *(const B16*)(qrow + d0);
            fq[c].q[1] = *(const B16*)(qrow + d0 + 16);
        }
    }

    v8f acc[8];
    #pragma unroll
    for (int j = 0; j < 8; j++) acc[j] = vzero8();
    float mrun[8], lrun[8];
    #pragma unroll
    for (int i = 0; i < 8; i++) { mrun[i] = -3.0e38f; lrun[i] = 0.f; }

    const int nkc = qb / 2 + 1;   // #32-key chunks needed (causal)
    for (int kc = 0; kc < nkc; kc++) {
        const int k0 = kc * 32;

        // ---- kick off V chunk staging (overlaps with QK^T + softmax) ----
#if HAVE_TDM
        asm volatile("s_wait_dscnt 0x0" ::: "memory");  // prior PV reads done
        tdm_load_2d(lds_off_of(&ldsV[0]), vtbase + k0,
                    (unsigned)S_, (unsigned)(NKV_ * D_),
                    (unsigned long long)S_, 32, 128);
#else
        for (int r = lane; r < 128; r += 32) {
            const unsigned short* src = vtbase + (size_t)r * S_ + k0;
            #pragma unroll
            for (int c = 0; c < 4; c++)
                *(B16*)&ldsV[r * 32 + c * 8] = *(const B16*)(src + c * 8);
        }
#endif

        // ---- scores for two 16-key tiles ----
        v8f sc[2];
        #pragma unroll
        for (int t = 0; t < 2; t++) {
            const int kt = k0 + t * 16;
            v8f s = vzero8();
            const unsigned short* krow = kbase + (size_t)(kt + lh) * kstride;
            // prefetch next chunk's K rows
            {
                int kn = (kt + 32 < S_) ? (kt + 32) : kt;
                __builtin_prefetch(kbase + (size_t)(kn + lh) * kstride, 0, 1);
            }
            #pragma unroll
            for (int c = 0; c < 4; c++) {
                FragBF fk;                    // B-frag: lane = key col, 16 d's
                int d0 = c * 32 + hi * 8;
                fk.q[0] = *(const B16*)(krow + d0);
                fk.q[1] = *(const B16*)(krow + d0 + 16);
                s = __builtin_amdgcn_wmma_f32_16x16x32_bf16(
                        false, fq[c].v, false, fk.v, (short)0, s, false, false);
            }
            const int kcol = kt + lh;
            const float mneg = amask[(size_t)b * S_ + kcol] ? 0.f : -1.0e30f;
            #pragma unroll
            for (int i = 0; i < 8; i++) {
                const int qrow = qb * 16 + i + hi * 8;
                float v = s[i] * SCALE_ + mneg;
                if (kcol > qrow) v = -1.0e30f;     // causal
                s[i] = v;
            }
            sc[t] = s;
        }

        // ---- online softmax: row reductions across 16-lane half-waves ----
        float mx[8];
        #pragma unroll
        for (int i = 0; i < 8; i++) mx[i] = fmaxf(sc[0][i], sc[1][i]);
        #pragma unroll
        for (int off = 8; off >= 1; off >>= 1)
            #pragma unroll
            for (int i = 0; i < 8; i++)
                mx[i] = fmaxf(mx[i], __shfl_xor(mx[i], off, 32));

        float alpha[8], rsum[8];
        #pragma unroll
        for (int i = 0; i < 8; i++) {
            float mnew = fmaxf(mrun[i], mx[i]);
            alpha[i] = __expf(mrun[i] - mnew);
            mrun[i] = mnew;
            float p0 = __expf(sc[0][i] - mnew);
            float p1 = __expf(sc[1][i] - mnew);
            sc[0][i] = p0; sc[1][i] = p1;
            rsum[i] = p0 + p1;
        }
        #pragma unroll
        for (int off = 8; off >= 1; off >>= 1)
            #pragma unroll
            for (int i = 0; i < 8; i++)
                rsum[i] += __shfl_xor(rsum[i], off, 32);
        #pragma unroll
        for (int i = 0; i < 8; i++) lrun[i] = lrun[i] * alpha[i] + rsum[i];

        #pragma unroll
        for (int j = 0; j < 8; j++)
            #pragma unroll
            for (int i = 0; i < 8; i++) acc[j][i] *= alpha[i];

        // ---- P: C layout -> LDS [m][k] (bf16) ----
        #pragma unroll
        for (int t = 0; t < 2; t++)
            #pragma unroll
            for (int i = 0; i < 8; i++)
                ldsP[(i + hi * 8) * 32 + t * 16 + lh] = f2bf(sc[t][i]);

#if HAVE_TDM
        __builtin_amdgcn_s_wait_tensorcnt(0);   // V chunk landed
#endif
        __syncthreads();                        // order P stores vs loads

        // ---- PV: one A-frag (P, K=32), 8 B-frags from ldsV, 8 WMMAs ----
        FragBF fp;
        fp.q[0] = *(const B16*)&ldsP[lh * 32 + hi * 8];
        fp.q[1] = *(const B16*)&ldsP[lh * 32 + hi * 8 + 16];
        #pragma unroll
        for (int j = 0; j < 8; j++) {
            FragBF fv;
            const unsigned short* pv = &ldsV[(j * 16 + lh) * 32 + hi * 8];
            fv.q[0] = *(const B16*)(pv);
            fv.q[1] = *(const B16*)(pv + 16);
            acc[j] = __builtin_amdgcn_wmma_f32_16x16x32_bf16(
                        false, fp.v, false, fv.v, (short)0, acc[j], false, false);
        }
        __syncthreads();
    }

    // ---- normalize and write bf16 output [B,S,NH,D] ----
    unsigned short* orow = O + ((size_t)b * S_ + qb * 16) * qstride + (size_t)h * D_;
    #pragma unroll
    for (int j = 0; j < 8; j++)
        #pragma unroll
        for (int i = 0; i < 8; i++) {
            float val = acc[j][i] / lrun[i];
            orow[(size_t)(i + hi * 8) * qstride + j * 16 + lh] = f2bf(val);
        }
}

// ---------------------------------------------------------------------------
// Host launcher
// ---------------------------------------------------------------------------
extern "C" void kernel_launch(void* const* d_in, const int* in_sizes, int n_in,
                              void* d_out, int out_size, void* d_ws, size_t ws_size,
                              hipStream_t stream) {
    (void)in_sizes; (void)n_in; (void)out_size; (void)ws_size;

    const float* hs   = (const float*)d_in[0];
    const float* cosb = (const float*)d_in[1];
    const float* sinb = (const float*)d_in[2];
    const unsigned char* amask = (const unsigned char*)d_in[3];
    const float* wq = (const float*)d_in[4];
    const float* wk = (const float*)d_in[5];
    const float* wv = (const float*)d_in[6];
    const float* wo = (const float*)d_in[7];

    char* ws = (char*)d_ws;
    size_t off = 0;
    auto alloc = [&](size_t bytes) -> void* {
        void* p = ws + off;
        off += (bytes + 255) & ~(size_t)255;
        return p;
    };
    const size_t M = (size_t)B_ * S_;                 // 2048
    const int NQ = NH_ * D_;                          // 4096
    const int NK = NKV_ * D_;                         // 1024

    unsigned short* hs_b = (unsigned short*)alloc(M * H_ * 2);
    unsigned short* wqt  = (unsigned short*)alloc((size_t)NQ * H_ * 2); // [N][K]
    unsigned short* wkt  = (unsigned short*)alloc((size_t)NK * H_ * 2);
    unsigned short* wvt  = (unsigned short*)alloc((size_t)NK * H_ * 2);
    unsigned short* wot  = (unsigned short*)alloc((size_t)H_ * NQ * 2); // [H][NH*D]
    unsigned short* q_b  = (unsigned short*)alloc(M * NQ * 2);
    unsigned short* k_b  = (unsigned short*)alloc(M * NK * 2);
    unsigned short* v_b  = (unsigned short*)alloc(M * NK * 2);
    unsigned short* vt   = (unsigned short*)alloc(M * NK * 2);          // [B][NK][S]
    unsigned short* ao_b = (unsigned short*)alloc(M * NQ * 2);

    // hidden states: straight convert (A operand, row-major)
    {
        int n8 = (int)(M * H_ / 8);
        cvt_bf16_v8<<<(n8 + 255) / 256, 256, 0, stream>>>(hs, hs_b, n8);
    }
    // weights: fused convert + transpose -> [N][K] bf16
    dim3 tb(32, 8);
    cvt_transpose_kernel<<<dim3(NQ / 32, H_ / 32), tb, 0, stream>>>(wq, wqt, H_, NQ);
    cvt_transpose_kernel<<<dim3(NK / 32, H_ / 32), tb, 0, stream>>>(wk, wkt, H_, NK);
    cvt_transpose_kernel<<<dim3(NK / 32, H_ / 32), tb, 0, stream>>>(wv, wvt, H_, NK);
    cvt_transpose_kernel<<<dim3(H_ / 32, NQ / 32), tb, 0, stream>>>(wo, wot, NQ, H_);

    // QKV projections (bf16 out)
    gemm_bf16_kernel<true><<<dim3(NQ / 128, (int)M / 64), 128, 0, stream>>>(
        hs_b, wqt, q_b, (int)M, NQ, H_);
    gemm_bf16_kernel<true><<<dim3(NK / 128, (int)M / 64), 128, 0, stream>>>(
        hs_b, wkt, k_b, (int)M, NK, H_);
    gemm_bf16_kernel<true><<<dim3(NK / 128, (int)M / 64), 128, 0, stream>>>(
        hs_b, wvt, v_b, (int)M, NK, H_);

    // RoPE on Q and K (in place)
    {
        int tq = B_ * S_ * NH_ * (D_ / 2);
        rope_kernel<<<(tq + 255) / 256, 256, 0, stream>>>(q_b, cosb, sinb, NH_);
        int tk = B_ * S_ * NKV_ * (D_ / 2);
        rope_kernel<<<(tk + 255) / 256, 256, 0, stream>>>(k_b, cosb, sinb, NKV_);
    }

    // V transpose per batch: [S][NK] -> [NK][S]
    for (int b = 0; b < B_; b++)
        transpose_bf16_kernel<<<dim3(NK / 32, S_ / 32), tb, 0, stream>>>(
            v_b + (size_t)b * S_ * NK, vt + (size_t)b * NK * S_, S_, NK);

    // Flash attention: one wave per (b, head, 16-row q tile)
    flash_kernel<<<B_ * NH_ * (S_ / 16), 32, 0, stream>>>(q_b, k_b, vt, amask, ao_b);

    // Output projection (fp32 out -> d_out)
    gemm_bf16_kernel<false><<<dim3(H_ / 128, (int)M / 64), 128, 0, stream>>>(
        ao_b, wot, (float*)d_out, (int)M, H_, NQ);
}
